// CrossModalFusion_74311524155429
// MI455X (gfx1250) — matrix operand order repaired and checked by reading, not compiled, not run on hardware
//
#include <hip/hip_runtime.h>

// ---------------------------------------------------------------------------
// CrossModalFusion on gfx1250 (MI455X).
// softmax over a single key == 1  =>  attention ctx == V-projection.
//   1) convert img/txt + live weights (Wv = rows [2d,3d) of Wi, Wo) to bf16
//   2) v      = bf16( X_kv @ Wv^T + bv )        (WMMA bf16, f32 accum)
//   3) ctx    = v @ Wo^T + bo                   (WMMA bf16, f32 out)
//   4) fused residual + layernorm + concat + elementwise product
// GEMM: 128x128 block tile, BK=64, double-buffered LDS, 8 waves/WG,
//       16 v_wmma_f32_16x16x32_bf16 per wave per K-iteration.
// Tile fill: GLOBAL_LOAD_ASYNC_TO_LDS_B128 (ASYNCcnt) -- zero staging VGPRs,
//            DMA overlaps WMMA; fallback: transient VGPR staging.
// ---------------------------------------------------------------------------

#if defined(__has_builtin)
#  if __has_builtin(__builtin_amdgcn_global_load_async_to_lds_b128)
#    define USE_ASYNC_LDS 1
#  endif
#endif
#ifndef USE_ASYNC_LDS
#  define USE_ASYNC_LDS 0
#endif

#if USE_ASYNC_LDS
#  if __has_builtin(__builtin_amdgcn_s_wait_asynccnt)
#    define WAIT_ASYNC0() __builtin_amdgcn_s_wait_asynccnt(0)
#  else
#    define WAIT_ASYNC0() asm volatile("s_wait_asynccnt 0x0" ::: "memory")
#  endif
typedef __attribute__((__vector_size__(4 * sizeof(int)))) int v4i;
typedef __attribute__((address_space(1))) v4i as1_v4i;   // global memory
typedef __attribute__((address_space(3))) v4i as3_v4i;   // LDS
#endif

typedef __attribute__((ext_vector_type(16))) __bf16 v16bf;
typedef __attribute__((ext_vector_type(8)))  float  v8f;

struct Frag32 { uint4 lo; uint4 hi; };   // 32 bytes -> bit_cast to v16bf

__device__ __forceinline__ unsigned short f2bf(float f) {
    unsigned u = __builtin_bit_cast(unsigned, f);
    u += 0x7FFFu + ((u >> 16) & 1u);          // round-to-nearest-even
    return (unsigned short)(u >> 16);
}

// pack two f32 -> two bf16 in one dword: low16 = bf16(x), high16 = bf16(y)
__device__ __forceinline__ unsigned pk2bf(float x, float y) {
    unsigned ux = __builtin_bit_cast(unsigned, x);
    unsigned uy = __builtin_bit_cast(unsigned, y);
    ux += 0x7FFFu + ((ux >> 16) & 1u);
    uy += 0x7FFFu + ((uy >> 16) & 1u);
    return __builtin_amdgcn_perm(uy, ux, 0x07060302u);  // both high halves
}

// ------------------------- f32 -> bf16 conversion --------------------------
__global__ __launch_bounds__(256)
void cvt_f32_bf16_kernel(const float* __restrict__ in,
                         unsigned* __restrict__ out, int n4) {
    int p = blockIdx.x * 256 + threadIdx.x;
    if (p < n4) {
        float4 f = ((const float4*)in)[p];
        uint2 o;
        o.x = pk2bf(f.x, f.y);
        o.y = pk2bf(f.z, f.w);
        ((uint2*)out)[p] = o;
    }
}

// ------------------------------ GEMM ---------------------------------------
// C[M,1024] = A[M,1024] @ W[1024,1024]^T + bias
// A, W bf16 row-major (W is [N,K]); C f32 or bf16 per template.
#define GK    1024
#define BM    128
#define BN    128
#define BK    64
#define NIT   (GK / BK)     // 16
#define LDSS  72            // padded row stride in ushorts (144 B, 16B-aligned)

template<bool OUT_BF16>
__global__ __launch_bounds__(256)
void gemm_wmma_bf16(const unsigned short* __restrict__ A,
                    const unsigned short* __restrict__ W,
                    const float* __restrict__ bias,
                    void* __restrict__ Cptr) {
    __shared__ __align__(16) unsigned short lds_a[2][BM * LDSS];
    __shared__ __align__(16) unsigned short lds_b[2][BN * LDSS];

    const int tid  = threadIdx.x;
    const int lane = tid & 31;
    const int wave = tid >> 5;
    const int wm   = wave & 1;     // 2 waves over M (64 rows each)
    const int wn   = wave >> 1;    // 4 waves over N (32 cols each)
    const int l15  = lane & 15;
    const int lhi  = lane >> 4;

    const int n0 = blockIdx.x * BN;    // N tile: fast-varying -> A strip shared
    const int m0 = blockIdx.y * BM;

    const int lr = tid >> 1;       // tile row handled by this thread (0..127)
    const int lh = tid & 1;        // which 32-element half of the 64-wide row

    const size_t arow = (size_t)(m0 + lr) * GK + lh * 32;
    const size_t brow = (size_t)(n0 + lr) * GK + lh * 32;

    v8f zero = {};
    v8f acc[4][2];
#pragma unroll
    for (int i = 0; i < 4; ++i)
#pragma unroll
        for (int j = 0; j < 2; ++j) acc[i][j] = zero;

    auto compute = [&](int buf) {
#pragma unroll
        for (int ks = 0; ks < 2; ++ks) {
            v16bf av[4], bv[2];
            const int kbA = ks * 32 + lhi * 8;   // A: {kb..kb+8, kb+16..kb+24}
#pragma unroll
            for (int i = 0; i < 4; ++i) {
                int r = wm * 64 + i * 16 + l15;
                Frag32 f;
                f.lo = *(const uint4*)&lds_a[buf][r * LDSS + kbA];
                f.hi = *(const uint4*)&lds_a[buf][r * LDSS + kbA + 16];
                av[i] = __builtin_bit_cast(v16bf, f);
            }
            const int kbB = ks * 32 + lhi * 8;   // B: {kb..kb+8, kb+16..kb+24}
#pragma unroll
            for (int j = 0; j < 2; ++j) {
                int n = wn * 32 + j * 16 + l15;
                Frag32 f;
                f.lo = *(const uint4*)&lds_b[buf][n * LDSS + kbB];
                f.hi = *(const uint4*)&lds_b[buf][n * LDSS + kbB + 16];
                bv[j] = __builtin_bit_cast(v16bf, f);
            }
#pragma unroll
            for (int i = 0; i < 4; ++i)
#pragma unroll
                for (int j = 0; j < 2; ++j)
                    acc[i][j] = __builtin_amdgcn_wmma_f32_16x16x32_bf16(
                        false, av[i], false, bv[j], (short)0, acc[i][j],
                        false, false);
        }
    };

#if USE_ASYNC_LDS
    // ---- async DMA global -> LDS: no staging VGPRs, overlaps WMMA ----
    auto async_load = [&](int buf, int kt) {
        const unsigned short* ga = A + arow + kt;
        const unsigned short* gb = W + brow + kt;
        unsigned short* la = &lds_a[buf][lr * LDSS + lh * 32];
        unsigned short* lb = &lds_b[buf][lr * LDSS + lh * 32];
#pragma unroll
        for (int i = 0; i < 4; ++i)
            __builtin_amdgcn_global_load_async_to_lds_b128(
                (as1_v4i*)(ga + 8 * i), (as3_v4i*)(la + 8 * i), 0, 0);
#pragma unroll
        for (int i = 0; i < 4; ++i)
            __builtin_amdgcn_global_load_async_to_lds_b128(
                (as1_v4i*)(gb + 8 * i), (as3_v4i*)(lb + 8 * i), 0, 0);
    };

    async_load(0, 0);
#pragma unroll 1
    for (int it = 0; it < NIT; ++it) {
        const int cur = it & 1;
        WAIT_ASYNC0();                 // own loads for lds[cur] have landed
        __syncthreads();               // all waves: cur ready, nxt unreferenced
        if (it + 1 < NIT) async_load(cur ^ 1, (it + 1) * BK);  // DMA || WMMA
        compute(cur);
    }
#else
    // ---- fallback: transient VGPR staging (short live ranges, no spill) ----
    auto stage = [&](int buf, int kt) {
        const uint4* sa = (const uint4*)(A + arow + kt);
        const uint4* sb = (const uint4*)(W + brow + kt);
        uint4 ua[4], ub[4];
#pragma unroll
        for (int i = 0; i < 4; ++i) ua[i] = sa[i];
#pragma unroll
        for (int i = 0; i < 4; ++i) ub[i] = sb[i];
        uint4* pa = (uint4*)&lds_a[buf][lr * LDSS + lh * 32];
        uint4* pb = (uint4*)&lds_b[buf][lr * LDSS + lh * 32];
#pragma unroll
        for (int i = 0; i < 4; ++i) pa[i] = ua[i];
#pragma unroll
        for (int i = 0; i < 4; ++i) pb[i] = ub[i];
    };

    stage(0, 0);
    __syncthreads();
#pragma unroll 1
    for (int it = 0; it < NIT; ++it) {
        const int cur = it & 1;
        if (it + 1 < NIT) stage(cur ^ 1, (it + 1) * BK);  // writes idle buffer
        compute(cur);
        __syncthreads();               // nxt fully written; cur reads done
    }
#endif

    // ---- epilogue: bias add + store ----
#pragma unroll
    for (int i = 0; i < 4; ++i) {
#pragma unroll
        for (int j = 0; j < 2; ++j) {
            int   n     = n0 + wn * 32 + j * 16 + l15;
            float bval  = bias[n];
            int   mbase = m0 + wm * 64 + i * 16 + lhi * 8;
#pragma unroll
            for (int r = 0; r < 8; ++r) {
                float  val = acc[i][j][r] + bval;
                size_t off = (size_t)(mbase + r) * GK + n;
                if (OUT_BF16) ((unsigned short*)Cptr)[off] = f2bf(val);
                else          ((float*)Cptr)[off] = val;
            }
        }
    }
}

// --------------- fused residual + LN + concat + product --------------------
__global__ __launch_bounds__(256)
void fuse_ln_kernel(const float* __restrict__ img, const float* __restrict__ ctxi,
                    const float* __restrict__ txt, const float* __restrict__ ctxt,
                    const float* __restrict__ gi, const float* __restrict__ bi,
                    const float* __restrict__ gt, const float* __restrict__ bt,
                    float* __restrict__ out) {
    __shared__ float4 sm[256];
    const int b = blockIdx.x;
    const int t = threadIdx.x;
    const size_t row = (size_t)b * 1024;

    float4 a  = ((const float4*)(img  + row))[t];
    float4 c  = ((const float4*)(ctxi + row))[t];
    float4 a2 = ((const float4*)(txt  + row))[t];
    float4 c2 = ((const float4*)(ctxt + row))[t];

    float xi[4] = {a.x + c.x,  a.y + c.y,  a.z + c.z,  a.w + c.w};
    float xt[4] = {a2.x + c2.x, a2.y + c2.y, a2.z + c2.z, a2.w + c2.w};

    float si = 0.f, ssi = 0.f, st = 0.f, sst = 0.f;
#pragma unroll
    for (int k = 0; k < 4; ++k) {
        si += xi[k]; ssi += xi[k] * xi[k];
        st += xt[k]; sst += xt[k] * xt[k];
    }
    sm[t] = make_float4(si, ssi, st, sst);
    __syncthreads();
    for (int s = 128; s > 0; s >>= 1) {
        if (t < s) {
            float4 o = sm[t + s];
            sm[t].x += o.x; sm[t].y += o.y; sm[t].z += o.z; sm[t].w += o.w;
        }
        __syncthreads();
    }
    float4 tot = sm[0];
    const float inv_d = 1.0f / 1024.0f;
    float mui = tot.x * inv_d, vari = tot.y * inv_d - mui * mui;
    float mut = tot.z * inv_d, vart = tot.w * inv_d - mut * mut;
    float ri = rsqrtf(vari + 1e-5f);
    float rt = rsqrtf(vart + 1e-5f);

    float4 g1 = ((const float4*)gi)[t], b1 = ((const float4*)bi)[t];
    float4 g2 = ((const float4*)gt)[t], b2 = ((const float4*)bt)[t];
    float gi4[4] = {g1.x, g1.y, g1.z, g1.w}, bi4[4] = {b1.x, b1.y, b1.z, b1.w};
    float gt4[4] = {g2.x, g2.y, g2.z, g2.w}, bt4[4] = {b2.x, b2.y, b2.z, b2.w};

    float oi[4], ot[4], op[4];
#pragma unroll
    for (int k = 0; k < 4; ++k) {
        oi[k] = (xi[k] - mui) * ri * gi4[k] + bi4[k];
        ot[k] = (xt[k] - mut) * rt * gt4[k] + bt4[k];
        op[k] = oi[k] * ot[k];
    }
    float* orow = out + (size_t)b * 3072;
    ((float4*)(orow       ))[t] = make_float4(oi[0], oi[1], oi[2], oi[3]);
    ((float4*)(orow + 1024))[t] = make_float4(ot[0], ot[1], ot[2], ot[3]);
    ((float4*)(orow + 2048))[t] = make_float4(op[0], op[1], op[2], op[3]);
}

// ---------------------------------------------------------------------------
extern "C" void kernel_launch(void* const* d_in, const int* in_sizes, int n_in,
                              void* d_out, int out_size, void* d_ws, size_t ws_size,
                              hipStream_t stream) {
    const float* img   = (const float*)d_in[0];
    const float* txt   = (const float*)d_in[1];
    const float* Wi_it = (const float*)d_in[2];
    const float* bi_it = (const float*)d_in[3];
    const float* Wo_it = (const float*)d_in[4];
    const float* bo_it = (const float*)d_in[5];
    const float* Wi_ti = (const float*)d_in[6];
    const float* bi_ti = (const float*)d_in[7];
    const float* Wo_ti = (const float*)d_in[8];
    const float* bo_ti = (const float*)d_in[9];
    const float* g_img = (const float*)d_in[10];
    const float* b_img = (const float*)d_in[11];
    const float* g_txt = (const float*)d_in[12];
    const float* b_txt = (const float*)d_in[13];
    float* out = (float*)d_out;

    const int d = 1024;
    const int B = in_sizes[0] / d;          // 32768
    const size_t dd = (size_t)d * d;
    const size_t bd = (size_t)B * d;

    // workspace layout (~456 MiB)
    unsigned short* wv_it  = (unsigned short*)d_ws;     // 4 x 2 MiB weights
    unsigned short* wo_it  = wv_it + dd;
    unsigned short* wv_ti  = wo_it + dd;
    unsigned short* wo_ti  = wv_ti + dd;
    unsigned short* xb_img = wo_ti + dd;                // B*d bf16 (64 MiB)
    unsigned short* xb_txt = xb_img + bd;               // B*d bf16 (64 MiB)
    unsigned short* vbuf   = xb_txt + bd;               // B*d bf16, reused
    float* ctx_img = (float*)(vbuf + bd);               // B*d f32 (128 MiB)
    float* ctx_txt = ctx_img + bd;                      // B*d f32 (128 MiB)

    dim3 blk(256);

    // 1) bf16 conversions: weights (only V-proj rows [2d,3d) of Wi are live)
    {
        int n4 = (int)(dd / 4);
        dim3 g((n4 + 255) / 256);
        cvt_f32_bf16_kernel<<<g, blk, 0, stream>>>(Wi_it + 2 * dd, (unsigned*)wv_it, n4);
        cvt_f32_bf16_kernel<<<g, blk, 0, stream>>>(Wo_it,          (unsigned*)wo_it, n4);
        cvt_f32_bf16_kernel<<<g, blk, 0, stream>>>(Wi_ti + 2 * dd, (unsigned*)wv_ti, n4);
        cvt_f32_bf16_kernel<<<g, blk, 0, stream>>>(Wo_ti,          (unsigned*)wo_ti, n4);
    }
    // ... and activations
    {
        int n4 = (int)(bd / 4);
        dim3 g((n4 + 255) / 256);
        cvt_f32_bf16_kernel<<<g, blk, 0, stream>>>(img, (unsigned*)xb_img, n4);
        cvt_f32_bf16_kernel<<<g, blk, 0, stream>>>(txt, (unsigned*)xb_txt, n4);
    }

    dim3 gg(d / BN, B / BM);   // x = N tiles (8, fast) ; y = M tiles (256)
    // img branch: ctx uses x_kv = txt with it-weights
    gemm_wmma_bf16<true ><<<gg, blk, 0, stream>>>(xb_txt, wv_it, bi_it + 2 * d, vbuf);
    gemm_wmma_bf16<false><<<gg, blk, 0, stream>>>(vbuf,   wo_it, bo_it,         ctx_img);
    // txt branch: ctx uses x_kv = img with ti-weights
    gemm_wmma_bf16<true ><<<gg, blk, 0, stream>>>(xb_img, wv_ti, bi_ti + 2 * d, vbuf);
    gemm_wmma_bf16<false><<<gg, blk, 0, stream>>>(vbuf,   wo_ti, bo_ti,         ctx_txt);

    // fused residual + LN + concat + product
    fuse_ln_kernel<<<B, 256, 0, stream>>>(img, ctx_img, txt, ctx_txt,
                                          g_img, b_img, g_txt, b_txt, out);
}
